// Gaussian_MultiHeadSA2_52183852646740
// MI455X (gfx1250) — matrix operand order repaired and checked by reading, not compiled, not run on hardware
//
#include <hip/hip_runtime.h>
#include <cstdint>

// Gaussian multi-head self-attention for MI455X (gfx1250), wave32 + WMMA f16.
//   B=8, T=1024, D=512, E=8 heads, D_E=64.
// Pipeline (4 kernels, all f16 WMMA with f32 accumulate):
//   k_qkv    : Q/K/V = X @ W^T + b, stored f16 in torch-permute head layout
//   k_scores : P' = exp(|QK^T/32 * punish|) (masked rows -> 1), f16 to ws
//              (fits CDNA5's 192MB L2). K tiles double-buffered into LDS via
//              GLOBAL_LOAD_ASYNC_TO_LDS_B128 + s_wait_asynccnt when available.
//   k_ctx    : ctx = invden * (P' @ V), scattered to mergedT[b][d][t] f16
//   k_out    : out = mergedT^T @ Wo^T + bo + inputQ (f32 result)

#define B_   8
#define T_   1024
#define D_   512
#define E_   8
#define DE_  64

typedef __attribute__((ext_vector_type(16))) _Float16 v16h;
typedef __attribute__((ext_vector_type(8)))  _Float16 v8h;
typedef __attribute__((ext_vector_type(8)))  float    v8f;
typedef __attribute__((ext_vector_type(4)))  int      v4i;

#if defined(__AMDGCN__) && defined(__has_builtin)
#if __has_builtin(__builtin_amdgcn_global_load_async_to_lds_b128) && \
    __has_builtin(__builtin_amdgcn_s_wait_asynccnt)
#define ASYNC_LDS 1
#endif
#endif

#ifdef ASYNC_LDS
typedef __attribute__((address_space(1))) v4i* as1_v4i;   // global (AS1)
typedef __attribute__((address_space(3))) v4i* as3_v4i;   // LDS (AS3)
static __device__ __forceinline__ void async_copy16(const void* g, void* l){
  // signature (probe-confirmed): (v4i AS1*, v4i AS3*, imm offset, imm cpol)
  __builtin_amdgcn_global_load_async_to_lds_b128(
      (as1_v4i)(unsigned long long)(uintptr_t)g,
      (as3_v4i)(unsigned int)(uintptr_t)l, 0, 0);
}
#endif

static __device__ __forceinline__ v16h cat16(v8h a, v8h b){
  return __builtin_shufflevector(a, b, 0,1,2,3,4,5,6,7,8,9,10,11,12,13,14,15);
}
// A operand (16x32 f16): lane = M%16; halves: K = {0..7,16..23} + 8*(lane>=16)
static __device__ __forceinline__ v16h frag_a(const _Float16* p, int stride, int lane){
  const _Float16* q = p + (lane & 15) * stride + ((lane & 16) ? 8 : 0);
  v8h lo = *(const v8h*)q;
  v8h hi = *(const v8h*)(q + 16);
  return cat16(lo, hi);
}
// B operand (32x16 f16): lane = N%16; halves: K = {0..7,16..23} + 8*(lane>=16)
static __device__ __forceinline__ v16h frag_b(const _Float16* p, int stride, int lane){
  const _Float16* q = p + (lane & 15) * stride + ((lane & 16) ? 8 : 0);
  v8h lo = *(const v8h*)q;
  v8h hi = *(const v8h*)(q + 16);
  return cat16(lo, hi);
}
static __device__ __forceinline__ v8f wmma16(v16h a, v16h b, v8f c){
  return __builtin_amdgcn_wmma_f32_16x16x32_f16(false, a, false, b, (short)0, c,
                                                false, false);
}
// pack two f16 (from f32) into one dword for ds_store_b32
static __device__ __forceinline__ unsigned pkh(float a, float b){
  union { _Float16 h[2]; unsigned u; } pk;
  pk.h[0] = (_Float16)a; pk.h[1] = (_Float16)b;
  return pk.u;
}
static __device__ __forceinline__ unsigned pkh16(_Float16 a, _Float16 b){
  union { _Float16 h[2]; unsigned u; } pk;
  pk.h[0] = a; pk.h[1] = b;
  return pk.u;
}

// ---------------- Kernel 1: QKV projections + head-split f16 store ----------
__global__ __launch_bounds__(256)
void k_qkv(const float* __restrict__ Xq, const float* __restrict__ Xk,
           const float* __restrict__ Xv,
           const float* __restrict__ Wq, const float* __restrict__ Wk,
           const float* __restrict__ Wv,
           const float* __restrict__ bq, const float* __restrict__ bk,
           const float* __restrict__ bv,
           _Float16* __restrict__ Qh, _Float16* __restrict__ Kh,
           _Float16* __restrict__ Vh)
{
  const float *X, *W, *bias; _Float16* dst;
  if (blockIdx.z == 0)      { X = Xq; W = Wq; bias = bq; dst = Qh; }
  else if (blockIdx.z == 1) { X = Xk; W = Wk; bias = bk; dst = Kh; }
  else                      { X = Xv; W = Wv; bias = bv; dst = Vh; }

  __shared__ _Float16 sA[128][40];   // 128 rows x 32 k  (+8 pad)
  __shared__ _Float16 sB[64][40];    // 64 cols x 32 k
  __shared__ _Float16 sCT[64][136];  // transposed C tile [n][m] (+8 pad)

  const int tid = threadIdx.x, lane = tid & 31, wid = tid >> 5;
  const int wm = wid & 3, wn = wid >> 2;
  const int M0 = blockIdx.y * 128;          // global row (b*T + t)
  const int n0 = blockIdx.x * 64;           // output feature block
  const int hi8 = (lane & 16) ? 8 : 0;

  v8f acc[2][2] = {};

  for (int k0 = 0; k0 < D_; k0 += 32){
    __syncthreads();
    for (int t = 0; t < 8; ++t){            // A: 2048 f32-pairs -> packed u32
      int idx = tid + t * 256;
      int m = idx >> 4, kp = idx & 15;
      float2 f = *(const float2*)&X[(size_t)(M0 + m) * D_ + k0 + 2 * kp];
      *(unsigned*)&sA[m][2 * kp] = pkh(f.x, f.y);
    }
    for (int t = 0; t < 4; ++t){            // B: 1024 f32-pairs
      int idx = tid + t * 256;
      int n = idx >> 4, kp = idx & 15;
      float2 f = *(const float2*)&W[(size_t)(n0 + n) * D_ + k0 + 2 * kp];
      *(unsigned*)&sB[n][2 * kp] = pkh(f.x, f.y);
    }
    __syncthreads();
    v16h a0 = frag_a(&sA[wm * 32][0],      40, lane);
    v16h a1 = frag_a(&sA[wm * 32 + 16][0], 40, lane);
    v16h b0 = frag_b(&sB[wn * 32][0],      40, lane);
    v16h b1 = frag_b(&sB[wn * 32 + 16][0], 40, lane);
    acc[0][0] = wmma16(a0, b0, acc[0][0]);
    acc[0][1] = wmma16(a0, b1, acc[0][1]);
    acc[1][0] = wmma16(a1, b0, acc[1][0]);
    acc[1][1] = wmma16(a1, b1, acc[1][1]);
  }

  for (int nt = 0; nt < 2; ++nt){
    int n = wn * 32 + nt * 16 + (lane & 15);
    float bn = bias[n0 + n];
    for (int mt = 0; mt < 2; ++mt){
      int m0 = wm * 32 + mt * 16 + hi8;
      for (int r = 0; r < 8; r += 2)        // packed ds_store_b32 (m contiguous)
        *(unsigned*)&sCT[n][m0 + r] =
            pkh(acc[mt][nt][r] + bn, acc[mt][nt][r + 1] + bn);
    }
  }
  __syncthreads();

  // torch permute(0,2,1).reshape head split:
  //   head = b*8 + (d>>6); i = (d&63)*16 + (t>>6); j = t&63
  const int b   = M0 >> 10;
  const int t0  = M0 & 1023;                 // aligned to 128
  const int head = b * E_ + blockIdx.x;      // d>>6 == blockIdx.x
  _Float16* dh = dst + (size_t)head * (T_ * DE_);
  for (int c = tid; c < 1024; c += 256){     // 8192 halves as half8 runs
    int dl = c >> 4, il = (c >> 3) & 1, j8 = c & 7;
    int i = dl * 16 + (t0 >> 6) + il;
    v8h v = *(const v8h*)&sCT[dl][il * 64 + j8 * 8];
    *(v8h*)&dh[i * 64 + j8 * 8] = v;
  }
}

// ---------------- Kernel 2: scores -> punish -> mask -> exp(|.|), row sums --
__global__ __launch_bounds__(256)
void k_scores(const _Float16* __restrict__ Qh, const _Float16* __restrict__ Kh,
              _Float16* __restrict__ Pp, float* __restrict__ invden,
              const float* __restrict__ theta,
              const long long* __restrict__ seq_len)
{
  __shared__ _Float16 sK[2][128][72];  // double-buffered K tile (+8 pad)
  __shared__ float sDen[128];

  const int tid = threadIdx.x, lane = tid & 31, wid = tid >> 5;
  const int wm = wid & 3, sw = wid >> 2;
  const int head = blockIdx.x;
  const int i0 = blockIdx.y * 128;
  const int bb = head >> 3;
  const int hi8 = (lane & 16) ? 8 : 0;

  const _Float16* Q = Qh + (size_t)head * (T_ * DE_);
  const _Float16* K = Kh + (size_t)head * (T_ * DE_);
  _Float16* P = Pp + (size_t)head * (T_ * T_) + (size_t)i0 * T_;

  const float th = theta[0];
  const float inv_t2 = 1.0f / (th * th);
  const long long sl = seq_len[bb];

  for (int t2 = tid; t2 < 128; t2 += 256) sDen[t2] = 0.f;

  // Q fragments for this wave's 32 rows, reused for every s-block
  v16h aQ[2][2];
  for (int mt = 0; mt < 2; ++mt)
    for (int kf = 0; kf < 2; ++kf)
      aQ[mt][kf] = frag_a(Q + (size_t)(i0 + wm * 32 + mt * 16) * DE_ + kf * 32,
                          DE_, lane);

  // 4 copy ops per wave per buffer; CDNA5 path issues them on the async unit
  auto issueK = [&](int buf, int s0){
    for (int t = 0; t < 4; ++t){
      int idx = tid + t * 256;             // 1024 16-byte chunks
      int row = idx >> 3, c8 = idx & 7;
#ifdef ASYNC_LDS
      async_copy16(&K[(size_t)(s0 + row) * DE_ + c8 * 8],
                   &sK[buf][row][c8 * 8]);
#else
      *(v8h*)&sK[buf][row][c8 * 8] =
          *(const v8h*)&K[(size_t)(s0 + row) * DE_ + c8 * 8];
#endif
    }
  };

  float rs[2][8] = {};
  issueK(0, 0);

  for (int sb = 0; sb < 8; ++sb){
    const int s0 = sb * 128;
    const int cur = sb & 1;
    __syncthreads();                       // everyone done with prev buffer
    if (sb + 1 < 8) issueK(cur ^ 1, s0 + 128);
#ifdef ASYNC_LDS
    if (sb + 1 < 8) __builtin_amdgcn_s_wait_asynccnt(4);  // cur's 4 done
    else            __builtin_amdgcn_s_wait_asynccnt(0);
#endif
    __syncthreads();                       // cur buffer visible to all waves

    for (int nt = 0; nt < 4; ++nt){
      v8f acc0 = {}, acc1 = {};
      for (int kf = 0; kf < 2; ++kf){
        v16h bkf = frag_b(&sK[cur][sw * 64 + nt * 16][kf * 32], 72, lane);
        acc0 = wmma16(aQ[0][kf], bkf, acc0);
        acc1 = wmma16(aQ[1][kf], bkf, acc1);
      }
      int s_g = s0 + sw * 64 + nt * 16 + (lane & 15);
      for (int mt = 0; mt < 2; ++mt){
        v8f a = mt ? acc1 : acc0;
        for (int r = 0; r < 8; ++r){
          int i_l = wm * 32 + mt * 16 + r + hi8;
          float v = a[r] * 0.03125f;                 // 1/sqrt(T)
          float dt = (float)(i0 + i_l - s_g);
          v *= __expf(-dt * dt * inv_t2);            // Gaussian punish
          if ((long long)(i0 + i_l) >= sl) v = 0.f;  // masked row -> uniform
          v = __expf(fabsf(v));                      // abs-softmax numerator
          rs[mt][r] += v;
          P[(size_t)i_l * T_ + s_g] = (_Float16)v;
        }
      }
    }
  }

  // reduce row sums: 16-lane group holds one row's 16 columns
  for (int mt = 0; mt < 2; ++mt)
    for (int r = 0; r < 8; ++r)
      for (int msk = 1; msk < 16; msk <<= 1)
        rs[mt][r] += __shfl_xor(rs[mt][r], msk, 32);
  __syncthreads();
  if ((lane & 15) == 0){
    for (int mt = 0; mt < 2; ++mt)
      for (int r = 0; r < 8; ++r)
        atomicAdd(&sDen[wm * 32 + mt * 16 + r + hi8], rs[mt][r]);
  }
  __syncthreads();
  if (tid < 128)
    invden[(size_t)head * T_ + i0 + tid] = 1.0f / sDen[tid];
}

// ---------------- Kernel 3: ctx = invden * (P' @ V) -> mergedT[b][d][t] -----
__global__ __launch_bounds__(256)
void k_ctx(const _Float16* __restrict__ Pp, const _Float16* __restrict__ Vh,
           const float* __restrict__ invden, _Float16* __restrict__ mergedT)
{
  __shared__ _Float16 sVT[64][136];  // V^T: [j][s-in-block] (+8 pad)
  __shared__ _Float16 sCtx[128][72]; // ctx tile [i][j]

  const int tid = threadIdx.x, lane = tid & 31, wid = tid >> 5;
  const int wm = wid & 3, wn = wid >> 2;
  const int head = blockIdx.x;
  const int i0 = blockIdx.y * 128;
  const int bb = head >> 3, e = head & 7;
  const int hi8 = (lane & 16) ? 8 : 0;

  const _Float16* P = Pp + (size_t)head * (T_ * T_) + (size_t)i0 * T_;
  const _Float16* V = Vh + (size_t)head * (T_ * DE_);

  v8f acc[2][2] = {};

  for (int sb = 0; sb < 8; ++sb){
    const int s0 = sb * 128;
    __syncthreads();
    for (int t = 0; t < 2; ++t){           // transpose by row pairs, b32 stores
      int idx = tid + t * 256;             // 512 pair-chunks
      int rp = idx >> 3, c8 = idx & 7;
      v8h v0 = *(const v8h*)&V[(size_t)(s0 + 2 * rp    ) * DE_ + c8 * 8];
      v8h v1 = *(const v8h*)&V[(size_t)(s0 + 2 * rp + 1) * DE_ + c8 * 8];
      for (int q = 0; q < 8; ++q)
        *(unsigned*)&sVT[c8 * 8 + q][2 * rp] = pkh16(v0[q], v1[q]);
    }
    __syncthreads();
    if (sb + 1 < 8){                       // warm next P' rows (L2 -> L0)
      __builtin_prefetch(P + (size_t)(wm * 32 + (lane & 15)) * T_ + s0 + 128, 0, 0);
      __builtin_prefetch(P + (size_t)(wm * 32 + 16 + (lane & 15)) * T_ + s0 + 128, 0, 0);
    }
    for (int kf = 0; kf < 4; ++kf){
      v16h a0 = frag_a(P + (size_t)(wm * 32) * T_ + s0 + kf * 32,      T_, lane);
      v16h a1 = frag_a(P + (size_t)(wm * 32 + 16) * T_ + s0 + kf * 32, T_, lane);
      v16h b0 = frag_b(&sVT[wn * 32][kf * 32],      136, lane);
      v16h b1 = frag_b(&sVT[wn * 32 + 16][kf * 32], 136, lane);
      acc[0][0] = wmma16(a0, b0, acc[0][0]);
      acc[0][1] = wmma16(a0, b1, acc[0][1]);
      acc[1][0] = wmma16(a1, b0, acc[1][0]);
      acc[1][1] = wmma16(a1, b1, acc[1][1]);
    }
  }

  float sc[2][8];
  for (int mt = 0; mt < 2; ++mt)
    for (int r = 0; r < 8; ++r)
      sc[mt][r] = invden[(size_t)head * T_ + i0 + wm * 32 + mt * 16 + r + hi8];

  for (int mt = 0; mt < 2; ++mt)
    for (int nt = 0; nt < 2; ++nt){
      int n = wn * 32 + nt * 16 + (lane & 15);
      int m0 = wm * 32 + mt * 16 + hi8;
      for (int r = 0; r < 8; ++r)          // [m][n]: r spans rows, keep b16
        sCtx[m0 + r][n] = (_Float16)(acc[mt][nt][r] * sc[mt][r]);
    }
  __syncthreads();

  // merged inverse map: d = e*64 + (i>>4), t = (i&15)*64 + j
  _Float16* MT = mergedT + (size_t)bb * (D_ * T_);
  for (int c = tid; c < 1024; c += 256){
    int i_l = c >> 3, j8 = c & 7;
    int i = i0 + i_l;
    int d = e * 64 + (i >> 4);
    int t = (i & 15) * 64 + j8 * 8;
    *(v8h*)&MT[(size_t)d * T_ + t] = *(const v8h*)&sCtx[i_l][j8 * 8];
  }
}

// ---------------- Kernel 4: out = merged @ Wo^T + bo + inputQ ---------------
__global__ __launch_bounds__(256)
void k_out(const _Float16* __restrict__ mergedT, const float* __restrict__ Wo,
           const float* __restrict__ bo, const float* __restrict__ inQ,
           float* __restrict__ out)
{
  __shared__ _Float16 sA[128][40];
  __shared__ _Float16 sB[64][40];

  const int tid = threadIdx.x, lane = tid & 31, wid = tid >> 5;
  const int wm = wid & 3, wn = wid >> 2;
  const int M0 = blockIdx.y * 128;
  const int n0 = blockIdx.x * 64;
  const int bb = M0 >> 10, t0 = M0 & 1023;
  const int hi8 = (lane & 16) ? 8 : 0;
  const _Float16* MT = mergedT + (size_t)bb * (D_ * T_);

  v8f acc[2][2] = {};
  for (int k0 = 0; k0 < D_; k0 += 32){
    __syncthreads();
    {                                       // A transpose by k pairs
      int c = tid;                          // 256 pair-chunks
      int kp = c >> 4, m8 = c & 15;
      v8h v0 = *(const v8h*)&MT[(size_t)(k0 + 2 * kp    ) * T_ + t0 + m8 * 8];
      v8h v1 = *(const v8h*)&MT[(size_t)(k0 + 2 * kp + 1) * T_ + t0 + m8 * 8];
      for (int q = 0; q < 8; ++q)
        *(unsigned*)&sA[m8 * 8 + q][2 * kp] = pkh16(v0[q], v1[q]);
    }
    for (int t = 0; t < 4; ++t){            // B: f32 pairs -> packed u32
      int idx = tid + t * 256;
      int n = idx >> 4, kp = idx & 15;
      float2 f = *(const float2*)&Wo[(size_t)(n0 + n) * D_ + k0 + 2 * kp];
      *(unsigned*)&sB[n][2 * kp] = pkh(f.x, f.y);
    }
    __syncthreads();
    v16h a0 = frag_a(&sA[wm * 32][0],      40, lane);
    v16h a1 = frag_a(&sA[wm * 32 + 16][0], 40, lane);
    v16h b0 = frag_b(&sB[wn * 32][0],      40, lane);
    v16h b1 = frag_b(&sB[wn * 32 + 16][0], 40, lane);
    acc[0][0] = wmma16(a0, b0, acc[0][0]);
    acc[0][1] = wmma16(a0, b1, acc[0][1]);
    acc[1][0] = wmma16(a1, b0, acc[1][0]);
    acc[1][1] = wmma16(a1, b1, acc[1][1]);
  }

  for (int nt = 0; nt < 2; ++nt){
    int col = n0 + wn * 32 + nt * 16 + (lane & 15);
    float bn = bo[col];
    for (int mt = 0; mt < 2; ++mt)
      for (int r = 0; r < 8; ++r){
        int m = M0 + wm * 32 + mt * 16 + r + hi8;
        size_t off = (size_t)m * D_ + col;
        out[off] = acc[mt][nt][r] + bn + inQ[off];
      }
  }
}

// ---------------------------------------------------------------------------
extern "C" void kernel_launch(void* const* d_in, const int* in_sizes, int n_in,
                              void* d_out, int out_size, void* d_ws, size_t ws_size,
                              hipStream_t stream)
{
  (void)in_sizes; (void)n_in; (void)out_size; (void)ws_size;
  const float* inputQ = (const float*)d_in[0];
  const float* inputK = (const float*)d_in[1];
  const float* inputV = (const float*)d_in[2];
  const float* Wq = (const float*)d_in[3];
  const float* bq = (const float*)d_in[4];
  const float* Wk = (const float*)d_in[5];
  const float* bk = (const float*)d_in[6];
  const float* Wv = (const float*)d_in[7];
  const float* bv = (const float*)d_in[8];
  const float* Wo = (const float*)d_in[9];
  const float* bo = (const float*)d_in[10];
  const float* theta = (const float*)d_in[11];
  const long long* seq_len = (const long long*)d_in[12];  // jnp.int64
  float* out = (float*)d_out;

  // workspace layout (bytes): Qh 8M | Kh 8M | Vh 8M | P' 128M | invden 1M | mergedT 8M
  char* w = (char*)d_ws;
  _Float16* Qh = (_Float16*)(w);
  _Float16* Kh = (_Float16*)(w + (size_t)8   * 1024 * 1024);
  _Float16* Vh = (_Float16*)(w + (size_t)16  * 1024 * 1024);
  _Float16* Pp = (_Float16*)(w + (size_t)24  * 1024 * 1024);
  float*    Dn = (float*)   (w + (size_t)152 * 1024 * 1024);
  _Float16* Mt = (_Float16*)(w + (size_t)153 * 1024 * 1024);

  k_qkv   <<<dim3(8, 64, 3), dim3(256), 0, stream>>>(inputQ, inputK, inputV,
                                                     Wq, Wk, Wv, bq, bk, bv,
                                                     Qh, Kh, Vh);
  k_scores<<<dim3(64, 8),    dim3(256), 0, stream>>>(Qh, Kh, Pp, Dn, theta, seq_len);
  k_ctx   <<<dim3(64, 8),    dim3(256), 0, stream>>>(Pp, Vh, Dn, Mt);
  k_out   <<<dim3(8, 64),    dim3(256), 0, stream>>>(Mt, Wo, bo, inputQ, out);
}
